// MpnnGGNN_3624952398581
// MI455X (gfx1250) — compile-verified, matrix operand order checked by reading, not run on hardware
//
#include <hip/hip_runtime.h>
#include <math.h>

typedef __attribute__((ext_vector_type(16))) _Float16     v16h;
typedef __attribute__((ext_vector_type(8)))  float        v8f;
typedef __attribute__((ext_vector_type(4)))  unsigned int v4u;

constexpr int BATCH   = 32;
constexpr int NNODE   = 128;
constexpr int IND     = 64;
constexpr int HDIM    = 128;
constexpr int NLAB    = 2;
constexpr int TGT     = 32;
constexpr int NLAYERS = 3;
constexpr int STR     = 136;   // row stride (halves) for A-consumed LDS; 68 dwords -> 16B-aligned rows, b128 loads
constexpr int BST     = 136;   // column stride (halves) for B-consumed (column-major) LDS

union FragB { unsigned int u[8]; v16h h; };

// ---------------- WMMA fragment helpers (wave32, 16x16x32 f16 -> f32) ----------------

__device__ __forceinline__ v8f wmma_f16(v16h a, v16h b, v8f c) {
  // emits v_wmma_f32_16x16x32_f16
  return __builtin_amdgcn_wmma_f32_16x16x32_f16(false, a, false, b, (short)0, c, false, false);
}

// A fragment (16x32) from row-major f16 LDS (stride STR): element(m,k) = base[(row0+m)*STR + k0+k]
// lane (g,m) needs halves k0+8g+{0..7} and k0+16+8g+{0..7}: two aligned b128 loads
__device__ __forceinline__ v16h ldA_lds(const _Float16* base, int row0, int k0, int lane) {
  const int g = lane >> 4, m = lane & 15;
  const _Float16* p = base + (row0 + m) * STR + k0 + 8 * g;
  FragB f;
  *(v4u*)&f.u[0] = *(const v4u*)p;
  *(v4u*)&f.u[4] = *(const v4u*)(p + 16);
  return f.h;
}

// B fragment (32x16) from column-major f16 LDS (stride BST): element(k,n) = base[(n)*BST + k]
// identical two-b128 pattern with n as the "row"
__device__ __forceinline__ v16h ldB_col(const _Float16* base, int k0, int n0, int lane) {
  const int g = lane >> 4, n = lane & 15;
  const _Float16* p = base + (n0 + n) * BST + k0 + 8 * g;
  FragB f;
  *(v4u*)&f.u[0] = *(const v4u*)p;
  *(v4u*)&f.u[4] = *(const v4u*)(p + 16);
  return f.h;
}

// K index for half-pair j (0..7) within a lane group g (ISA 16-bit A layout); for the scalar gather below
__device__ __forceinline__ int kmap(int j, int g) { return (j < 4 ? 2 * j : 2 * j + 8) + 8 * g; }

// A fragment of A_l^T: element(m=v, k=w) = am_b[(w*N + v)*L + l]  (0/1 values, f16-exact)
__device__ __forceinline__ v16h ldA_adjT(const float* __restrict__ amb, int l, int v0, int k0, int lane) {
  const int g = lane >> 4, m = lane & 15;
  const int v = v0 + m;
  v16h a;
#pragma unroll
  for (int j = 0; j < 8; ++j) {
    const int k = kmap(j, g);
    a[2 * j]     = (_Float16)amb[(size_t)((k0 + k)     * NNODE + v) * NLAB + l];
    a[2 * j + 1] = (_Float16)amb[(size_t)((k0 + k + 1) * NNODE + v) * NLAB + l];
  }
  return a;
}

// store C/D accumulator to row-major f16 LDS (stride STR): row = row0 + r + 8g, col = col0 + (lane&15)
__device__ __forceinline__ void stCD_row(_Float16* base, int row0, int col0, int lane, v8f c) {
  const int g = lane >> 4, n = lane & 15;
#pragma unroll
  for (int r = 0; r < 8; ++r)
    base[(row0 + r + 8 * g) * STR + col0 + n] = (_Float16)c[r];
}

// store C/D accumulator to column-major f16 LDS: rows r+8g for fixed column are 8 contiguous
// halves -> pack to one aligned b128 store
__device__ __forceinline__ void stCD_col(_Float16* base, int row0, int col0, int lane, v8f c) {
  const int g = lane >> 4, n = lane & 15;
  union { v4u u; _Float16 hh[8]; } pk;
#pragma unroll
  for (int r = 0; r < 8; ++r) pk.hh[r] = (_Float16)c[r];
  *(v4u*)(base + (col0 + n) * BST + row0 + 8 * g) = pk.u;
}

// ---------------- kernel: one workgroup per graph ----------------

__global__ __launch_bounds__(256) void ggnn_kernel(
    const float* __restrict__ h_in, const float* __restrict__ am, const int* __restrict__ g_size,
    const float* __restrict__ W_edge, const float* __restrict__ W_ih, const float* __restrict__ W_hh,
    const float* __restrict__ b_ih, const float* __restrict__ b_hh,
    const float* __restrict__ Wi, const float* __restrict__ bi,
    const float* __restrict__ Wj, const float* __restrict__ bj,
    float* __restrict__ out)
{
  __shared__ alignas(16) _Float16 sh_h[NNODE * STR];          // h_t (row-major, A-consumed)
  __shared__ alignas(16) _Float16 sh_w[NNODE * STR];          // scratch: We^T/X (col-major) -> m (row-major)
  __shared__ alignas(16) _Float16 sh_wih[3 * HDIM * BST];     // W_ih^T blocks [r,z,n] (col-major, B-ready)
  __shared__ alignas(16) _Float16 sh_whh[3 * HDIM * BST];     // W_hh^T blocks [r,z,n] (col-major, B-ready)
  __shared__ float sh_res[TGT];
  __shared__ float sh_real[NNODE];

  const int b    = blockIdx.x;
  const int tid  = threadIdx.x;
  const int lane = tid & 31;
  const int wv   = tid >> 5;               // wave id == row-tile index (0..7)
  const int v0   = wv * 16;
  const int gsz  = g_size[b];
  const float* hin_b = h_in + (size_t)b * NNODE * IND;
  const float* am_b  = am   + (size_t)b * NNODE * NNODE * NLAB;

  // h_t = [h_in, zeros]  (row-major)
  for (int p = tid; p < NNODE * HDIM; p += 256) {
    const int row = p >> 7, col = p & 127;
    const float v = (col < IND) ? hin_b[row * IND + col] : 0.0f;
    sh_h[row * STR + col] = (_Float16)v;
  }
  // stage GRU weights once per block: element(k, n) -> col-major buf[n*BST + k] (contiguous writes)
  for (int p = tid; p < 3 * HDIM * HDIM; p += 256) {
    const int row = p >> 7;                 // gate-output index 0..383
    const int k   = p & 127;                // input (K) index
    const int blk = row >> 7;               // r/z/n block
    const int n   = row & 127;
    const size_t idx = (size_t)blk * HDIM * BST + (size_t)n * BST + k;
    sh_wih[idx] = (_Float16)W_ih[p];
    sh_whh[idx] = (_Float16)W_hh[p];
  }
  __syncthreads();

  v8f zero = {};

  for (int layer = 0; layer < NLAYERS; ++layer) {
    v8f macc[8];
#pragma unroll
    for (int t = 0; t < 8; ++t) macc[t] = zero;

    for (int l = 0; l < NLAB; ++l) {
      // ---- stage We_l^T into sh_w col-major: element(k=i, n=o) = W_edge[l][o*H+i] -> sh_w[o*BST+i]
      const float* We = W_edge + (size_t)l * HDIM * HDIM;
      for (int p = tid; p < HDIM * HDIM; p += 256) {
        const int o = p >> 7, i = p & 127;
        sh_w[o * BST + i] = (_Float16)We[p];
      }
      __syncthreads();

      // ---- X_l = H (N x H) * We_l^T (H x H), into registers
      v8f xacc[8];
#pragma unroll
      for (int t = 0; t < 8; ++t) xacc[t] = zero;
      for (int kt = 0; kt < 4; ++kt) {
        v16h a = ldA_lds(sh_h, v0, kt * 32, lane);
#pragma unroll
        for (int t = 0; t < 8; ++t) {
          v16h bf = ldB_col(sh_w, kt * 32, t * 16, lane);
          xacc[t] = wmma_f16(a, bf, xacc[t]);
        }
      }
      __syncthreads();                       // We reads complete
#pragma unroll
      for (int t = 0; t < 8; ++t) stCD_col(sh_w, v0, t * 16, lane, xacc[t]);  // X overwrites We
      __syncthreads();

      // ---- m += A_l^T (N x N) * X_l (N x H)
      for (int kt = 0; kt < 4; ++kt) {
        v16h a = ldA_adjT(am_b, l, v0, kt * 32, lane);
#pragma unroll
        for (int t = 0; t < 8; ++t) {
          v16h bf = ldB_col(sh_w, kt * 32, t * 16, lane);
          macc[t] = wmma_f16(a, bf, macc[t]);
        }
      }
      __syncthreads();                       // X reads done before next overwrite
    }

    // ---- stage m into sh_w row-major (A-operand for the gate GEMMs)
#pragma unroll
    for (int t = 0; t < 8; ++t) stCD_row(sh_w, v0, t * 16, lane, macc[t]);
    __syncthreads();

    // preload A fragments for the gate GEMMs (depend only on kstep)
    v16h Amf[4], Ahf[4];
#pragma unroll
    for (int kt = 0; kt < 4; ++kt) {
      Amf[kt] = ldA_lds(sh_w, v0, kt * 32, lane);
      Ahf[kt] = ldA_lds(sh_h, v0, kt * 32, lane);
    }

    // ---- GRU gates: ir/iz/in = m*W_ih^T (blocks), hr/hz/hn = h*W_hh^T (blocks)
    v8f hnew[8];
    for (int t = 0; t < 8; ++t) {
      v8f ir = zero, iz = zero, inn = zero, hr = zero, hz = zero, hn = zero;
#pragma unroll
      for (int kt = 0; kt < 4; ++kt) {
        const int k0 = kt * 32, n0 = t * 16;
        v16h Br = ldB_col(sh_wih,                   k0, n0, lane);
        v16h Bz = ldB_col(sh_wih +     HDIM * BST,  k0, n0, lane);
        v16h Bn = ldB_col(sh_wih + 2 * HDIM * BST,  k0, n0, lane);
        ir  = wmma_f16(Amf[kt], Br, ir);
        iz  = wmma_f16(Amf[kt], Bz, iz);
        inn = wmma_f16(Amf[kt], Bn, inn);
        v16h Cr = ldB_col(sh_whh,                   k0, n0, lane);
        v16h Cz = ldB_col(sh_whh +     HDIM * BST,  k0, n0, lane);
        v16h Cn = ldB_col(sh_whh + 2 * HDIM * BST,  k0, n0, lane);
        hr = wmma_f16(Ahf[kt], Cr, hr);
        hz = wmma_f16(Ahf[kt], Cz, hz);
        hn = wmma_f16(Ahf[kt], Cn, hn);
      }
      const int g = lane >> 4, n = lane & 15;
      const int o = t * 16 + n;
      const float bir = b_ih[o],            bhr = b_hh[o];
      const float biz = b_ih[HDIM + o],     bhz = b_hh[HDIM + o];
      const float bin = b_ih[2 * HDIM + o], bhn = b_hh[2 * HDIM + o];
      v8f hv;
#pragma unroll
      for (int r = 0; r < 8; ++r) {
        const int row  = v0 + r + 8 * g;
        const float ho = (float)sh_h[row * STR + o];
        const float rr = 1.0f / (1.0f + __expf(-(ir[r] + bir + hr[r] + bhr)));
        const float zz = 1.0f / (1.0f + __expf(-(iz[r] + biz + hz[r] + bhz)));
        const float nn = tanhf(inn[r] + bin + rr * (hn[r] + bhn));
        const float hm = (1.0f - zz) * nn + zz * ho;
        hv[r] = (row < gsz) ? hm : 0.0f;     // node mask
      }
      hnew[t] = hv;
    }
    __syncthreads();                          // all sh_h/sh_w reads complete
#pragma unroll
    for (int t = 0; t < 8; ++t) stCD_row(sh_h, v0, t * 16, lane, hnew[t]);
    __syncthreads();
  }

  // ---------------- readout: sum_v real * sigmoid(i([h,hin])) * j(h), then log_softmax ----------------
  for (int p = tid; p < TGT; p += 256) sh_res[p] = 0.0f;
  for (int p = tid; p < NNODE; p += 256) {
    float s = 0.0f;
    for (int i = 0; i < IND; ++i) s += hin_b[p * IND + i];
    sh_real[p] = (s > 0.0f) ? 1.0f : 0.0f;
  }
  __syncthreads();
  {
    const int t = tid & 31;
    float acc = 0.0f;
    for (int v = tid >> 5; v < NNODE; v += 8) {
      float gsum = bi[t], vsum = bj[t];
      for (int o = 0; o < HDIM; ++o) {
        const float hv = (float)sh_h[v * STR + o];
        gsum += hv * Wi[o * TGT + t];
        vsum += hv * Wj[o * TGT + t];
      }
      for (int i = 0; i < IND; ++i)
        gsum += hin_b[v * IND + i] * Wi[(HDIM + i) * TGT + t];
      const float gate = 1.0f / (1.0f + __expf(-gsum));
      acc += sh_real[v] * gate * vsum;
    }
    atomicAdd(&sh_res[t], acc);
  }
  __syncthreads();
  if (tid == 0) {
    float mx = sh_res[0];
    for (int t = 1; t < TGT; ++t) mx = fmaxf(mx, sh_res[t]);
    float s = 0.0f;
    for (int t = 0; t < TGT; ++t) s += __expf(sh_res[t] - mx);
    const float lse = mx + __logf(s);
    for (int t = 0; t < TGT; ++t) out[b * TGT + t] = sh_res[t] - lse;
  }
}

extern "C" void kernel_launch(void* const* d_in, const int* in_sizes, int n_in,
                              void* d_out, int out_size, void* d_ws, size_t ws_size,
                              hipStream_t stream) {
  (void)in_sizes; (void)n_in; (void)out_size; (void)d_ws; (void)ws_size;
  ggnn_kernel<<<dim3(BATCH), dim3(256), 0, stream>>>(
      (const float*)d_in[0],  // h_in
      (const float*)d_in[1],  // am
      (const int*)  d_in[2],  // g_size
      (const float*)d_in[3],  // W_edge
      (const float*)d_in[4],  // W_ih
      (const float*)d_in[5],  // W_hh
      (const float*)d_in[6],  // b_ih
      (const float*)d_in[7],  // b_hh
      (const float*)d_in[8],  // Wi
      (const float*)d_in[9],  // bi
      (const float*)d_in[10], // Wj
      (const float*)d_in[11], // bj
      (float*)d_out);
}
